// KANBlock_84748294684697
// MI455X (gfx1250) — matrix operand order repaired and checked by reading, not compiled, not run on hardware
//
#include <hip/hip_runtime.h>
#include <cmath>

typedef __attribute__((ext_vector_type(16))) _Float16 v16h;
typedef __attribute__((ext_vector_type(8)))  float    v8f;

// ---------------------------------------------------------------------------
// Problem constants: M=16384 tokens, layer1 256->512 (K1=1792),
// layer2 512->256 (K2=3584), 6 spline bases + 1 silu feature per input dim.
// ---------------------------------------------------------------------------
#define TM 128
#define TN 128
#define TK 32
#define LDT 40   // padded LDS row stride (halves): 80B, multiple of 16B

// --------------------- CDNA5 async global->LDS copy ------------------------
#if defined(__AMDGCN__) && __has_builtin(__builtin_amdgcn_global_load_async_to_lds_b128)
#define KAN_ASYNC_LDS 1
#else
#define KAN_ASYNC_LDS 0
#endif

#if KAN_ASYNC_LDS
typedef int v4i_ __attribute__((vector_size(16)));
typedef __attribute__((address_space(1))) v4i_ gv4;   // global (__device__) AS
typedef __attribute__((address_space(3))) v4i_ lv4;   // LDS (__shared__) AS
static __device__ __forceinline__ void async_cp16(const void* g, void* l) {
    __builtin_amdgcn_global_load_async_to_lds_b128((gv4*)g, (lv4*)l, 0, 0);
}
static __device__ __forceinline__ void async_wait0() {
#if __has_builtin(__builtin_amdgcn_s_wait_asynccnt)
    __builtin_amdgcn_s_wait_asynccnt(0);
#else
    asm volatile("s_wait_asynccnt 0x0" ::: "memory");
#endif
}
#endif

// ---------------------------------------------------------------------------
// Build transposed, pre-scaled f16 weight matrix Wt[N][K]:
//   k <  6D : Wt[o][k] = coef[i][o][b] * scale_sp[i][o],  i=k/6, b=k%6
//   k >= 6D : Wt[o][k] = scale_base[k-6D][o]
// ---------------------------------------------------------------------------
__global__ __launch_bounds__(256) void kan_prep_wt(
    const float* __restrict__ coef, const float* __restrict__ ssp,
    const float* __restrict__ sbase, _Float16* __restrict__ Wt,
    int D, int N)
{
    const int K = 7 * D;
    size_t id = (size_t)blockIdx.x * 256 + threadIdx.x;
    if (id >= (size_t)N * K) return;
    int o = (int)(id / K);
    int k = (int)(id % K);
    float v;
    if (k < 6 * D) {
        int i = k / 6, b = k % 6;
        v = coef[((size_t)i * N + o) * 6 + b] * ssp[(size_t)i * N + o];
    } else {
        int i = k - 6 * D;
        v = sbase[(size_t)i * N + o];
    }
    Wt[id] = (_Float16)v;
}

// ---------------------------------------------------------------------------
// Feature builder: degree-3 Cox-de-Boor basis (6 values) + silu.
// F[n][i*6+b] = basis_b(x[n,i]);  F[n][6D+i] = silu(x[n,i]).
// ---------------------------------------------------------------------------
__global__ __launch_bounds__(256) void kan_features(
    const float* __restrict__ X, const float* __restrict__ grid,
    _Float16* __restrict__ F, int D, int Kf)
{
    size_t id = (size_t)blockIdx.x * 256 + threadIdx.x;   // id < M*D exactly
    int n = (int)(id / D);
    int i = (int)(id % D);
    float x = X[id];

    float g[10];
    #pragma unroll
    for (int j = 0; j < 10; ++j) g[j] = grid[i * 10 + j];

    float Bv[9];
    #pragma unroll
    for (int j = 0; j < 9; ++j)
        Bv[j] = (x >= g[j] && x < g[j + 1]) ? 1.0f : 0.0f;

    const float inv_h = 1.0f / (g[1] - g[0]);   // knots are uniform
    #pragma unroll
    for (int d = 1; d <= 3; ++d) {
        const float s = inv_h * ((d == 1) ? 1.0f : (d == 2) ? 0.5f : (1.0f / 3.0f));
        #pragma unroll
        for (int j = 0; j <= 8 - d; ++j) {
            float left  = (x - g[j]) * s;
            float right = (g[j + d + 1] - x) * s;
            Bv[j] = left * Bv[j] + right * Bv[j + 1];
        }
    }

    _Float16* Frow = F + (size_t)n * Kf + (size_t)i * 6;
    #pragma unroll
    for (int b = 0; b < 6; ++b) Frow[b] = (_Float16)Bv[b];

    float sx = x / (1.0f + __expf(-x));         // silu
    F[(size_t)n * Kf + (size_t)6 * D + i] = (_Float16)sx;
}

// ---------------------------------------------------------------------------
// WMMA GEMM:  Out[M][N] = nscale[o]*(F[M][K] @ Wt[N][K]^T) + nbias[o]
// 256 threads = 8 wave32 waves; 128x128 tile; K chunks of 32; LDS double
// buffer filled by GLOBAL_LOAD_ASYNC_TO_LDS_B128 (ASYNCcnt) when available.
// ---------------------------------------------------------------------------
__global__ __launch_bounds__(256, 1) void kan_gemm(
    const _Float16* __restrict__ F, const _Float16* __restrict__ Wt,
    const float* __restrict__ nscale, const float* __restrict__ nbias,
    float* __restrict__ Out, int N, int K)
{
    __shared__ __align__(16) _Float16 lds[2 * 2 * TM * LDT];

    const int tid  = threadIdx.x;
    const int wave = tid >> 5;
    const int lane = tid & 31;
    const int m0 = blockIdx.y * TM;
    const int n0 = blockIdx.x * TN;

    v8f acc[8];
    #pragma unroll
    for (int i = 0; i < 8; ++i)
        acc[i] = (v8f){0.f, 0.f, 0.f, 0.f, 0.f, 0.f, 0.f, 0.f};

    const int nk = K / TK;
    const int ldrow = tid >> 2;          // row within tile (two passes of 64)
    const int ldseg = tid & 3;           // 16B segment within the 64B row chunk

#if KAN_ASYNC_LDS
    // issue one tile's worth of async copies (A: 128x32 f16, B: 128x32 f16)
    auto stage_async = [&](int buf, int k0) {
        _Float16* A  = lds + buf * (2 * TM * LDT);
        _Float16* Bt = A + TM * LDT;
        #pragma unroll
        for (int j = 0; j < 2; ++j) {
            int r = ldrow + j * 64;
            async_cp16(F  + (size_t)(m0 + r) * K + k0 + ldseg * 8,
                       A  + r * LDT + ldseg * 8);
            async_cp16(Wt + (size_t)(n0 + r) * K + k0 + ldseg * 8,
                       Bt + r * LDT + ldseg * 8);
        }
    };
    stage_async(0, 0);
    async_wait0();
    __syncthreads();
#else
    uint4 regA[2], regB[2];
    {
        #pragma unroll
        for (int j = 0; j < 2; ++j) {
            int r = ldrow + j * 64;
            regA[j] = *(const uint4*)(F  + (size_t)(m0 + r) * K + ldseg * 8);
            regB[j] = *(const uint4*)(Wt + (size_t)(n0 + r) * K + ldseg * 8);
        }
        _Float16* A  = lds;
        _Float16* Bt = lds + TM * LDT;
        #pragma unroll
        for (int j = 0; j < 2; ++j) {
            int r = ldrow + j * 64;
            *(uint4*)(A  + r * LDT + ldseg * 8) = regA[j];
            *(uint4*)(Bt + r * LDT + ldseg * 8) = regB[j];
        }
    }
    __syncthreads();
#endif

    const int mrow  = wave * 16 + (lane & 15);
    const int koffA = (lane < 16) ? 0 : 16;   // A frag: lanes16-31 hold K16..31
    const int koffB = (lane < 16) ? 0 : 16;   // B frag: lanes16-31 hold K16..31
    const int colb  = lane & 15;

    for (int kc = 0; kc < nk; ++kc) {
        const int cur = kc & 1;

#if KAN_ASYNC_LDS
        if (kc + 1 < nk) stage_async(cur ^ 1, (kc + 1) * TK);
#else
        if (kc + 1 < nk) {
            const int k0 = (kc + 1) * TK;
            #pragma unroll
            for (int j = 0; j < 2; ++j) {
                int r = ldrow + j * 64;
                regA[j] = *(const uint4*)(F  + (size_t)(m0 + r) * K + k0 + ldseg * 8);
                regB[j] = *(const uint4*)(Wt + (size_t)(n0 + r) * K + k0 + ldseg * 8);
            }
        }
#endif

        const _Float16* A  = lds + cur * (2 * TM * LDT);
        const _Float16* Bt = A + TM * LDT;

        // A fragment: two b128 reads (K base..base+7, base+8..base+15)
        union frag { uint4 u[2]; v16h h; };
        frag ua;
        ua.u[0] = *(const uint4*)(A + mrow * LDT + koffA);
        ua.u[1] = *(const uint4*)(A + mrow * LDT + koffA + 8);

        // B fragments double-buffered in registers so ds loads overlap wmma
        frag ub[2];
        ub[0].u[0] = *(const uint4*)(Bt + colb * LDT + koffB);
        ub[0].u[1] = *(const uint4*)(Bt + colb * LDT + koffB + 8);
        #pragma unroll
        for (int ns = 0; ns < 8; ++ns) {
            if (ns < 7) {
                const int col = (ns + 1) * 16 + colb;
                ub[(ns + 1) & 1].u[0] = *(const uint4*)(Bt + col * LDT + koffB);
                ub[(ns + 1) & 1].u[1] = *(const uint4*)(Bt + col * LDT + koffB + 8);
            }
            acc[ns] = __builtin_amdgcn_wmma_f32_16x16x32_f16(
                false, ua.h, false, ub[ns & 1].h, (short)0, acc[ns], false, false);
        }

#if KAN_ASYNC_LDS
        if (kc + 1 < nk) async_wait0();
        __syncthreads();
#else
        if (kc + 1 < nk) {
            _Float16* An  = lds + (cur ^ 1) * (2 * TM * LDT);
            _Float16* Btn = An + TM * LDT;
            #pragma unroll
            for (int j = 0; j < 2; ++j) {
                int r = ldrow + j * 64;
                *(uint4*)(An  + r * LDT + ldseg * 8) = regA[j];
                *(uint4*)(Btn + r * LDT + ldseg * 8) = regB[j];
            }
        }
        __syncthreads();
#endif
    }

    // Epilogue: D layout -> VGPR r: M = r + (lane<16 ? 0 : 8), Ncol = lane%16
    const int rbase = (lane < 16) ? 0 : 8;
    #pragma unroll
    for (int ns = 0; ns < 8; ++ns) {
        const int col = n0 + ns * 16 + colb;
        const float s = nscale[col];
        const float b = nbias[col];
        #pragma unroll
        for (int r = 0; r < 8; ++r) {
            const int row = m0 + wave * 16 + rbase + r;
            Out[(size_t)row * N + col] = s * acc[ns][r] + b;
        }
    }
}

// ---------------------------------------------------------------------------
// Residual + LayerNorm(256) + exact GELU. One block per token (256 threads).
// ---------------------------------------------------------------------------
__global__ __launch_bounds__(256) void kan_ln_gelu(
    const float* __restrict__ H, const float* __restrict__ X,
    const float* __restrict__ gamma, const float* __restrict__ beta,
    float* __restrict__ Out)
{
    const int D = 256;
    const int n = blockIdx.x;
    const int o = threadIdx.x;
    const size_t idx = (size_t)n * D + o;

    float v = H[idx] + X[idx];

    __shared__ float red[8];
    const int w = o >> 5, l = o & 31;

    float s = v;
    #pragma unroll
    for (int off = 16; off > 0; off >>= 1) s += __shfl_down(s, off, 32);
    if (l == 0) red[w] = s;
    __syncthreads();
    float sum = 0.f;
    #pragma unroll
    for (int j = 0; j < 8; ++j) sum += red[j];
    const float mu = sum * (1.0f / 256.0f);
    __syncthreads();

    const float c = v - mu;
    float s2 = c * c;
    #pragma unroll
    for (int off = 16; off > 0; off >>= 1) s2 += __shfl_down(s2, off, 32);
    if (l == 0) red[w] = s2;
    __syncthreads();
    float sum2 = 0.f;
    #pragma unroll
    for (int j = 0; j < 8; ++j) sum2 += red[j];
    const float var = sum2 * (1.0f / 256.0f);

    const float r = rsqrtf(var + 1e-5f);
    const float y = c * r * gamma[o] + beta[o];
    Out[idx] = 0.5f * y * (1.0f + erff(y * 0.70710678118654752f));
}

// ---------------------------------------------------------------------------
extern "C" void kernel_launch(void* const* d_in, const int* in_sizes, int n_in,
                              void* d_out, int out_size, void* d_ws, size_t ws_size,
                              hipStream_t stream)
{
    const float* x     = (const float*)d_in[0];
    const float* grid1 = (const float*)d_in[1];
    const float* coef1 = (const float*)d_in[2];
    const float* sb1   = (const float*)d_in[3];
    const float* ssp1  = (const float*)d_in[4];
    const float* ns1   = (const float*)d_in[5];
    const float* nb1   = (const float*)d_in[6];
    const float* grid2 = (const float*)d_in[7];
    const float* coef2 = (const float*)d_in[8];
    const float* sb2   = (const float*)d_in[9];
    const float* ssp2  = (const float*)d_in[10];
    const float* ns2   = (const float*)d_in[11];
    const float* nb2   = (const float*)d_in[12];
    const float* gamma = (const float*)d_in[13];
    const float* beta  = (const float*)d_in[14];

    const int M  = 16384;
    const int D0 = 256, D1 = 512, D2 = 256;
    const int K1 = 7 * D0;   // 1792
    const int K2 = 7 * D1;   // 3584

    // Workspace layout (F2 reuses F1 region; h2 aliases h1 region).
    char* ws = (char*)d_ws;
    _Float16* Fm  = (_Float16*)ws;                                // max(M*K1, M*K2)
    size_t off    = (size_t)M * K2 * sizeof(_Float16);
    float*    h1  = (float*)(ws + off);
    off          += (size_t)M * D1 * sizeof(float);
    _Float16* Wt1 = (_Float16*)(ws + off);
    off          += (size_t)D1 * K1 * sizeof(_Float16);
    _Float16* Wt2 = (_Float16*)(ws + off);
    float*    h2  = h1;   // safe: h1 dead once features2 has produced F2

    kan_prep_wt<<<(D1 * K1 + 255) / 256, 256, 0, stream>>>(coef1, ssp1, sb1, Wt1, D0, D1);
    kan_prep_wt<<<(D2 * K2 + 255) / 256, 256, 0, stream>>>(coef2, ssp2, sb2, Wt2, D1, D2);

    kan_features<<<(M * D0) / 256, 256, 0, stream>>>(x, grid1, Fm, D0, K1);
    kan_gemm<<<dim3(D1 / TN, M / TM), 256, 0, stream>>>(Fm, Wt1, ns1, nb1, h1, D1, K1);

    kan_features<<<(M * D1) / 256, 256, 0, stream>>>(h1, grid2, Fm, D1, K2);
    kan_gemm<<<dim3(D2 / TN, M / TM), 256, 0, stream>>>(Fm, Wt2, ns2, nb2, h2, D2, K2);

    kan_ln_gelu<<<M, 256, 0, stream>>>(h2, x, gamma, beta, (float*)d_out);
}
